// TransformerBinaryClassifier_53145925320747
// MI455X (gfx1250) — matrix operand order repaired and checked by reading, not compiled, not run on hardware
//
#include <hip/hip_runtime.h>
#include <hip/hip_bf16.h>
#include <stdint.h>

typedef _Float16 h16;
typedef __attribute__((ext_vector_type(16))) _Float16 v16h;
typedef __attribute__((ext_vector_type(8)))  _Float16 v8h;
typedef __attribute__((ext_vector_type(8)))  float    v8f;

#define B_   32
#define S_   512
#define D_   256
#define H_   8
#define HD_  32
#define FF_  1024
#define NLAYER 4
#define MTOK (B_*S_)
#define EPS_ 1e-5f
#define LAM_ 1e-3f

// ---------------------------------------------------------------------------
// Cross-lane reductions within 16-lane halves via DPP16 ROW_XMASK
// (dpp_ctrl = 0x160 | xor_mask). Pure VALU: no LDS traffic, no DScnt waits.
// ---------------------------------------------------------------------------
template<int MASK>
__device__ __forceinline__ float dpp_xor(float v) {
  const int s = __builtin_amdgcn_update_dpp(0, __float_as_int(v),
                                            0x160 | MASK, 0xF, 0xF, true);
  return __int_as_float(s);
}
__device__ __forceinline__ float rowmax16(float v) {
  v = fmaxf(v, dpp_xor<1>(v));
  v = fmaxf(v, dpp_xor<2>(v));
  v = fmaxf(v, dpp_xor<4>(v));
  v = fmaxf(v, dpp_xor<8>(v));
  return v;
}
__device__ __forceinline__ float rowsum16(float v) {
  v += dpp_xor<1>(v);
  v += dpp_xor<2>(v);
  v += dpp_xor<4>(v);
  v += dpp_xor<8>(v);
  return v;
}

// ---------------------------------------------------------------------------
// Generic GEMM: C[M,N] = A[M,K] (f16) @ W[N,K]^T (f16) + bias, f32 accumulate
// via v_wmma_f32_16x16x32_f16. One wave (32 threads) computes a 16x64 tile.
// MODE 0: f32 out [M,N]     MODE 1: f16 out [M,N]
// MODE 2: f16 out + ReLU    MODE 3: f16 out transposed -> [B, N, Sdim]
// A-operand layout (16-bit A 16x32): lane m=l&15, hi=l>>4;
//   halves 0..15 -> k = 16*hi .. 16*hi+15 (matches B)
// B-operand layout (32x16): lane n=l&15 (+16 -> k+16); 16 contiguous halves.
// C layout: lane n=l&15, VGPR r -> row m = r + 8*(l>>4).
// ---------------------------------------------------------------------------
template<int MODE>
__global__ void gemm_xwT(const h16* __restrict__ A, const h16* __restrict__ W,
                         const float* __restrict__ bias, void* __restrict__ outp,
                         int M, int N, int K, int Sdim)
{
  (void)M; (void)Sdim;
  const int l  = threadIdx.x;
  const int ln = l & 15, hi = l >> 4;
  const int m0 = blockIdx.x * 16;
  const int n0 = blockIdx.y * 64;

  v8f acc0 = {}, acc1 = {}, acc2 = {}, acc3 = {};
  const h16* arow = A + (size_t)(m0 + ln) * K;
  const h16* w0 = W + (size_t)(n0 +  0 + ln) * K + hi * 16;
  const h16* w1 = W + (size_t)(n0 + 16 + ln) * K + hi * 16;
  const h16* w2 = W + (size_t)(n0 + 32 + ln) * K + hi * 16;
  const h16* w3 = W + (size_t)(n0 + 48 + ln) * K + hi * 16;

  for (int k0 = 0; k0 < K; k0 += 32) {
    const h16* ap = arow + k0 + 16 * hi;
    if (k0 + 32 < K) {
      __builtin_prefetch((const void*)(arow + k0 + 32), 0, 0);  // global_prefetch_b8
    }
    v8h alo = *(const v8h*)(ap);
    v8h aup = *(const v8h*)(ap + 8);
    v16h av = __builtin_shufflevector(alo, aup, 0,1,2,3,4,5,6,7,8,9,10,11,12,13,14,15);
    v16h b0 = *(const v16h*)(w0 + k0);
    v16h b1 = *(const v16h*)(w1 + k0);
    v16h b2 = *(const v16h*)(w2 + k0);
    v16h b3 = *(const v16h*)(w3 + k0);
    acc0 = __builtin_amdgcn_wmma_f32_16x16x32_f16(false, av, false, b0, (short)0, acc0, false, false);
    acc1 = __builtin_amdgcn_wmma_f32_16x16x32_f16(false, av, false, b1, (short)0, acc1, false, false);
    acc2 = __builtin_amdgcn_wmma_f32_16x16x32_f16(false, av, false, b2, (short)0, acc2, false, false);
    acc3 = __builtin_amdgcn_wmma_f32_16x16x32_f16(false, av, false, b3, (short)0, acc3, false, false);
  }

  const float bv0 = bias[n0 +  0 + ln];
  const float bv1 = bias[n0 + 16 + ln];
  const float bv2 = bias[n0 + 32 + ln];
  const float bv3 = bias[n0 + 48 + ln];

#pragma unroll
  for (int r = 0; r < 8; ++r) {
    const int m = m0 + r + 8 * hi;
    float v0 = acc0[r] + bv0, v1 = acc1[r] + bv1, v2 = acc2[r] + bv2, v3 = acc3[r] + bv3;
    if (MODE == 2) {
      v0 = fmaxf(v0, 0.f); v1 = fmaxf(v1, 0.f); v2 = fmaxf(v2, 0.f); v3 = fmaxf(v3, 0.f);
    }
    if (MODE == 0) {
      float* o = (float*)outp + (size_t)m * N + n0 + ln;
      o[0] = v0; o[16] = v1; o[32] = v2; o[48] = v3;
    } else if (MODE == 3) {
      const int b = m / Sdim, s = m - b * Sdim;
      h16* o = (h16*)outp;
      o[((size_t)b * N + n0 +  0 + ln) * Sdim + s] = (h16)v0;
      o[((size_t)b * N + n0 + 16 + ln) * Sdim + s] = (h16)v1;
      o[((size_t)b * N + n0 + 32 + ln) * Sdim + s] = (h16)v2;
      o[((size_t)b * N + n0 + 48 + ln) * Sdim + s] = (h16)v3;
    } else {
      h16* o = (h16*)outp + (size_t)m * N + n0 + ln;
      o[0] = (h16)v0; o[16] = (h16)v1; o[32] = (h16)v2; o[48] = (h16)v3;
    }
  }
}

// ---------------------------------------------------------------------------
// Flash attention: one wave per (b, h, 16-query tile). Online softmax over
// 32-key steps; P transposed C->A layout through LDS; V pre-transposed to
// [B, H*HD, S] so the P@V B-operand is contiguous along keys.
// ---------------------------------------------------------------------------
__global__ void attn_kernel(const h16* __restrict__ qh, const h16* __restrict__ kh,
                            const h16* __restrict__ vt, const uint8_t* __restrict__ mask,
                            h16* __restrict__ attnh)
{
  __shared__ __align__(32) h16 Pl[4][16][32];
  const int tid = threadIdx.x;
  const int w = tid >> 5, l = tid & 31;
  const int ln = l & 15, hi = l >> 4;
  const int task = blockIdx.x * 4 + w;
  const int qt = task & 31;          // S/16 = 32 query tiles
  const int h  = (task >> 5) & 7;
  const int b  = task >> 8;

  // Q fragment (A operand): rows = 16 queries, K = HD = 32
  const h16* qp = qh + ((size_t)(b * S_ + qt * 16 + ln)) * D_ + h * HD_ + 16 * hi;
  v8h qlo = *(const v8h*)(qp);
  v8h qup = *(const v8h*)(qp + 8);
  v16h qa = __builtin_shufflevector(qlo, qup, 0,1,2,3,4,5,6,7,8,9,10,11,12,13,14,15);

  float Mrow[8], Lrow[8];
#pragma unroll
  for (int r = 0; r < 8; ++r) { Mrow[r] = -1e30f; Lrow[r] = 0.f; }
  v8f o0 = {}, o1 = {};
  const float scale = 0.17677669529663687f; // 1/sqrt(HD)

  for (int kt = 0; kt < S_; kt += 32) {
    const h16* kp0 = kh + ((size_t)(b * S_ + kt +      ln)) * D_ + h * HD_ + hi * 16;
    const h16* kp1 = kh + ((size_t)(b * S_ + kt + 16 + ln)) * D_ + h * HD_ + hi * 16;
    v16h kb0 = *(const v16h*)kp0;
    v16h kb1 = *(const v16h*)kp1;
    if (kt + 32 < S_) {
      __builtin_prefetch((const void*)(kp0 + (size_t)32 * D_), 0, 0);
      __builtin_prefetch((const void*)(kp1 + (size_t)32 * D_), 0, 0);
    }
    v8f z = {};
    v8f s0 = __builtin_amdgcn_wmma_f32_16x16x32_f16(false, qa, false, kb0, (short)0, z, false, false);
    v8f s1 = __builtin_amdgcn_wmma_f32_16x16x32_f16(false, qa, false, kb1, (short)0, z, false, false);

    const float add0 = mask[b * S_ + kt +      ln] ? -1e30f : 0.f;
    const float add1 = mask[b * S_ + kt + 16 + ln] ? -1e30f : 0.f;

#pragma unroll
    for (int r = 0; r < 8; ++r) {
      float a0 = s0[r] * scale + add0;
      float a1 = s1[r] * scale + add1;
      float rm = rowmax16(fmaxf(a0, a1));          // DPP16 ROW_XMASK reduce
      float mn = fmaxf(Mrow[r], rm);
      float p0 = __expf(a0 - mn);
      float p1 = __expf(a1 - mn);
      float corr = __expf(Mrow[r] - mn);
      float rs = rowsum16(p0 + p1);                // DPP16 ROW_XMASK reduce
      Lrow[r] = Lrow[r] * corr + rs;
      Mrow[r] = mn;
      o0[r] = o0[r] * corr;
      o1[r] = o1[r] * corr;
      Pl[w][r + 8 * hi][ln]      = (h16)p0;   // transpose C-layout -> row-major
      Pl[w][r + 8 * hi][ln + 16] = (h16)p1;
    }
    asm volatile("s_wait_dscnt 0" ::: "memory");  // wave-local LDS RAW

    // reload P in A-operand layout [16 x 32]
    const h16* pp = &Pl[w][ln][0] + 16 * hi;
    v8h plo = *(const v8h*)(pp);
    v8h pup = *(const v8h*)(pp + 8);
    v16h pa = __builtin_shufflevector(plo, pup, 0,1,2,3,4,5,6,7,8,9,10,11,12,13,14,15);

    // V B-operands: vt[B, H*HD, S], contiguous along keys
    const h16* vp0 = vt + ((size_t)b * D_ + h * HD_ +  0 + ln) * S_ + kt + hi * 16;
    const h16* vp1 = vt + ((size_t)b * D_ + h * HD_ + 16 + ln) * S_ + kt + hi * 16;
    v16h vb0 = *(const v16h*)vp0;
    v16h vb1 = *(const v16h*)vp1;
    o0 = __builtin_amdgcn_wmma_f32_16x16x32_f16(false, pa, false, vb0, (short)0, o0, false, false);
    o1 = __builtin_amdgcn_wmma_f32_16x16x32_f16(false, pa, false, vb1, (short)0, o1, false, false);
  }

#pragma unroll
  for (int r = 0; r < 8; ++r) {
    const float inv = 1.f / Lrow[r];
    const size_t row = (size_t)(b * S_ + qt * 16 + r + 8 * hi);
    attnh[row * D_ + h * HD_ +  0 + ln] = (h16)(o0[r] * inv);
    attnh[row * D_ + h * HD_ + 16 + ln] = (h16)(o1[r] * inv);
  }
}

// ---------------------------------------------------------------------------
__global__ void ln_residual(const float* __restrict__ xin, const float* __restrict__ dlt,
                            const float* __restrict__ g, const float* __restrict__ be,
                            float* __restrict__ xout, h16* __restrict__ xhout)
{
  __shared__ float red[256];
  const int t = threadIdx.x;
  const size_t row = blockIdx.x;
  const float v = xin[row * D_ + t] + dlt[row * D_ + t];
  red[t] = v; __syncthreads();
  for (int s2 = 128; s2 > 0; s2 >>= 1) { if (t < s2) red[t] += red[t + s2]; __syncthreads(); }
  const float mu = red[0] / (float)D_;
  __syncthreads();
  const float d = v - mu;
  red[t] = d * d; __syncthreads();
  for (int s2 = 128; s2 > 0; s2 >>= 1) { if (t < s2) red[t] += red[t + s2]; __syncthreads(); }
  const float var = red[0] / (float)D_;
  const float y = d * rsqrtf(var + EPS_) * g[t] + be[t];
  xout[row * D_ + t] = y;
  xhout[row * D_ + t] = (h16)y;
}

__global__ void pool_bn(const float* __restrict__ x, const uint8_t* __restrict__ mask,
                        const float* __restrict__ bng, const float* __restrict__ bnb,
                        const float* __restrict__ bnm, const float* __restrict__ bnv,
                        float* __restrict__ pooled, float* __restrict__ out_mean)
{
  const int b = blockIdx.x, t = threadIdx.x;
  float acc = 0.f, cnt = 0.f;
  for (int s = 0; s < S_; ++s) {
    const float valid = mask[b * S_ + s] ? 0.f : 1.f;
    acc += x[((size_t)b * S_ + s) * D_ + t] * valid;
    cnt += valid;
  }
  const float m = acc / cnt;
  const float y = (m - bnm[t]) * rsqrtf(bnv[t] + EPS_) * bng[t] + bnb[t];
  pooled[b * D_ + t] = y;
  out_mean[b * D_ + t] = y;
}

__global__ void score_kernel(const float* __restrict__ pooled,
                             const float* __restrict__ nm, const float* __restrict__ ncov,
                             const float* __restrict__ am, const float* __restrict__ acov,
                             float* __restrict__ logits)
{
  __shared__ float mv[256];
  __shared__ float redn[256];
  __shared__ float reda[256];
  const int b = blockIdx.x, t = threadIdx.x;
  mv[t] = pooled[b * D_ + t];
  __syncthreads();
  float tn = 0.f, ta = 0.f;
  for (int j = 0; j < D_; ++j) {
    const float mj = mv[j];
    tn += ncov[(size_t)t * D_ + j] * mj;
    ta += acov[(size_t)t * D_ + j] * mj;
  }
  const float mi = mv[t];
  redn[t] = mi * nm[t] + 0.5f * LAM_ * mi * tn;
  reda[t] = mi * am[t] + 0.5f * LAM_ * mi * ta;
  __syncthreads();
  for (int s2 = 128; s2 > 0; s2 >>= 1) {
    if (t < s2) { redn[t] += redn[t + s2]; reda[t] += reda[t + s2]; }
    __syncthreads();
  }
  if (t == 0) logits[b] = 1.f / (1.f + __expf(-(reda[0] - redn[0])));
}

__global__ void f32_to_f16_k(const float* __restrict__ in, h16* __restrict__ out, int n)
{
  const int i = blockIdx.x * 256 + threadIdx.x;
  if (i < n) out[i] = (h16)in[i];
}

__global__ void copy_and_half_k(const float* __restrict__ in, float* __restrict__ of,
                                h16* __restrict__ oh, int n)
{
  const int i = blockIdx.x * 256 + threadIdx.x;
  if (i < n) { const float v = in[i]; of[i] = v; oh[i] = (h16)v; }
}

// ---------------------------------------------------------------------------
extern "C" void kernel_launch(void* const* d_in, const int* in_sizes, int n_in,
                              void* d_out, int out_size, void* d_ws, size_t ws_size,
                              hipStream_t stream) {
  (void)in_sizes; (void)n_in; (void)out_size; (void)ws_size;

  const float* raw  = (const float*)d_in[0];
  const float* gnn  = (const float*)d_in[1];
  const uint8_t* mask = (const uint8_t*)d_in[2];
  const float* Wq = (const float*)d_in[3];
  const float* Wk = (const float*)d_in[4];
  const float* Wv = (const float*)d_in[5];
  const float* bq = (const float*)d_in[6];
  const float* bk = (const float*)d_in[7];
  const float* bv = (const float*)d_in[8];
  const float* Wo = (const float*)d_in[9];
  const float* bo = (const float*)d_in[10];
  const float* W1 = (const float*)d_in[11];
  const float* b1 = (const float*)d_in[12];
  const float* W2 = (const float*)d_in[13];
  const float* b2 = (const float*)d_in[14];
  const float* g1 = (const float*)d_in[15];
  const float* be1 = (const float*)d_in[16];
  const float* g2 = (const float*)d_in[17];
  const float* be2 = (const float*)d_in[18];
  const float* bng = (const float*)d_in[19];
  const float* bnb = (const float*)d_in[20];
  const float* bnm = (const float*)d_in[21];
  const float* bnv = (const float*)d_in[22];
  const float* normal_mean  = (const float*)d_in[23];
  const float* normal_cov   = (const float*)d_in[24];
  const float* abnormal_mean = (const float*)d_in[25];
  const float* abnormal_cov  = (const float*)d_in[26];

  float* out = (float*)d_out;

  const size_t BSD = (size_t)B_ * S_ * D_;    // 4,194,304
  char* ws = (char*)d_ws;
  size_t off = 0;
  auto wsalloc = [&](size_t bytes) -> void* {
    void* p = (void*)(ws + off);
    off += (bytes + 255) & ~(size_t)255;
    return p;
  };
  float* x     = (float*)wsalloc(BSD * 4);
  h16*   xh    = (h16*)  wsalloc(BSD * 2);
  h16*   qh    = (h16*)  wsalloc(BSD * 2);
  h16*   khb   = (h16*)  wsalloc(BSD * 2);
  h16*   vt    = (h16*)  wsalloc(BSD * 2);   // [B, H*HD, S]
  h16*   attnh = (h16*)  wsalloc(BSD * 2);
  float* tmp   = (float*)wsalloc(BSD * 4);
  h16*   h1    = (h16*)  wsalloc((size_t)MTOK * FF_ * 2);
  h16*   wq16  = (h16*)  wsalloc((size_t)D_ * D_ * 2);
  h16*   wk16  = (h16*)  wsalloc((size_t)D_ * D_ * 2);
  h16*   wv16  = (h16*)  wsalloc((size_t)D_ * D_ * 2);
  h16*   wo16  = (h16*)  wsalloc((size_t)D_ * D_ * 2);
  h16*   w116  = (h16*)  wsalloc((size_t)FF_ * D_ * 2);
  h16*   w216  = (h16*)  wsalloc((size_t)FF_ * D_ * 2);
  float* pooled = (float*)wsalloc((size_t)B_ * D_ * 4);

  const int DD = D_ * D_, FD = FF_ * D_;
  f32_to_f16_k<<<DD / 256, 256, 0, stream>>>(Wq, wq16, DD);
  f32_to_f16_k<<<DD / 256, 256, 0, stream>>>(Wk, wk16, DD);
  f32_to_f16_k<<<DD / 256, 256, 0, stream>>>(Wv, wv16, DD);
  f32_to_f16_k<<<DD / 256, 256, 0, stream>>>(Wo, wo16, DD);
  f32_to_f16_k<<<FD / 256, 256, 0, stream>>>(W1, w116, FD);
  f32_to_f16_k<<<FD / 256, 256, 0, stream>>>(W2, w216, FD);

  // x = GNN_output (f32 + f16); raw -> f16 (staged in qh) for the V projection
  copy_and_half_k<<<(int)(BSD / 256), 256, 0, stream>>>(gnn, x, xh, (int)BSD);
  f32_to_f16_k<<<(int)(BSD / 256), 256, 0, stream>>>(raw, qh, (int)BSD);

  const dim3 gD(MTOK / 16, D_ / 64);   // N = 256
  const dim3 gF(MTOK / 16, FF_ / 64);  // N = 1024

  // V = raw @ Wv^T + bv, stored transposed [B, H*HD, S]
  gemm_xwT<3><<<gD, 32, 0, stream>>>(qh, wv16, bv, vt, MTOK, D_, D_, S_);

  for (int layer = 0; layer < NLAYER; ++layer) {
    gemm_xwT<1><<<gD, 32, 0, stream>>>(xh, wq16, bq, qh,  MTOK, D_, D_, 0);
    gemm_xwT<1><<<gD, 32, 0, stream>>>(xh, wk16, bk, khb, MTOK, D_, D_, 0);
    attn_kernel<<<(B_ * H_ * (S_ / 16)) / 4, 128, 0, stream>>>(qh, khb, vt, mask, attnh);
    gemm_xwT<0><<<gD, 32, 0, stream>>>(attnh, wo16, bo, tmp, MTOK, D_, D_, 0);
    ln_residual<<<MTOK, 256, 0, stream>>>(x, tmp, g1, be1, x, xh);
    gemm_xwT<2><<<gF, 32, 0, stream>>>(xh, w116, b1, h1, MTOK, FF_, D_, 0);
    gemm_xwT<0><<<gD, 32, 0, stream>>>(h1, w216, b2, tmp, MTOK, D_, FF_, 0);
    ln_residual<<<MTOK, 256, 0, stream>>>(x, tmp, g2, be2, x, xh);
  }

  pool_bn<<<B_, 256, 0, stream>>>(x, mask, bng, bnb, bnm, bnv, pooled, out + 32);
  score_kernel<<<B_, 256, 0, stream>>>(pooled, normal_mean, normal_cov,
                                       abnormal_mean, abnormal_cov, out);

  // pass-through outputs: normal_mean, normal_cov, abnormal_cov, abnormal_mean
  hipMemcpyAsync(out + 8224,   normal_mean,   (size_t)D_ * 4,       hipMemcpyDeviceToDevice, stream);
  hipMemcpyAsync(out + 8480,   normal_cov,    (size_t)D_ * D_ * 4,  hipMemcpyDeviceToDevice, stream);
  hipMemcpyAsync(out + 74016,  abnormal_cov,  (size_t)D_ * D_ * 4,  hipMemcpyDeviceToDevice, stream);
  hipMemcpyAsync(out + 139552, abnormal_mean, (size_t)D_ * 4,       hipMemcpyDeviceToDevice, stream);
}
